// P4Transformer_49830210568192
// MI455X (gfx1250) — matrix operand order repaired and checked by reading, not compiled, not run on hardware
//
#include <hip/hip_runtime.h>
#include <hip/hip_bf16.h>
#include <math.h>

// ---------------------------------------------------------------------------
// CDNA5 (gfx1250) implementation of P4Transformer forward.
// All dense contractions -> one WMMA f16 GEMM kernel (v_wmma_f32_16x16x32_f16),
// B-panel staged to LDS via the Tensor Data Mover when available.
// Irregular point ops (FPS / ball query / 3-NN interp) are wave32-native.
// ---------------------------------------------------------------------------

typedef __attribute__((ext_vector_type(16))) _Float16 v16h;
typedef __attribute__((ext_vector_type(8)))  float    v8f;
typedef __attribute__((ext_vector_type(4)))  unsigned int v4u;
typedef __attribute__((ext_vector_type(8)))  int      v8i;
typedef __attribute__((ext_vector_type(4)))  int      v4i;

enum {
  GF_RELU     = 1,
  GF_ROWSCALE = 2,
  GF_ROWBIAS  = 4,
  GF_COLBIAS  = 8,
  GF_ADD      = 16,
  GF_GELU     = 32,
  GF_TRANSB   = 64
};

#define KCHUNK 128

// ---------------------------------------------------------------------------
// Generic fused WMMA GEMM:  C[m,n] = act(alpha * sum_k A[m,k]*B[k,n] * g[m] ...)
// A row-major MxK (lda); B row-major KxN (ldb) or, with GF_TRANSB, NxK.
// One wave computes a 16x32 C strip (two 16x16 WMMA tiles, shared A fragment);
// 4 waves/block stack along M. B panel staged to LDS via TDM.
// ---------------------------------------------------------------------------
__global__ void __launch_bounds__(128)
gemm_wmma_kernel(const float* __restrict__ A, const float* __restrict__ B,
                 float* __restrict__ C,
                 const float* __restrict__ rs, const float* __restrict__ rb,
                 const float* __restrict__ cb, const float* __restrict__ addend,
                 int M, int N, int K, int lda, int ldb, int ldc, int ldadd,
                 long long sA, long long sB, long long sC, long long sAdd,
                 float alpha, int flags)
{
  const int lane = threadIdx.x & 31;
  const int wv   = threadIdx.x >> 5;
  const int tn   = blockIdx.x * 32;
  const int tm   = (blockIdx.y * 4 + wv) * 16;
  const int bz   = blockIdx.z;
  A += (long long)bz * sA;
  B += (long long)bz * sB;
  C += (long long)bz * sC;
  if (flags & GF_ADD) addend += (long long)bz * sAdd;

  const int lh = lane >> 4;   // lane-half: 0 or 1
  const int ll = lane & 15;

  __shared__ float smemB[KCHUNK * 32];

  v8f acc0 = {};
  v8f acc1 = {};

  const int  mrow  = tm + ll;
  const bool rowOK = (mrow < M);
  const float* Arow = A + (long long)(rowOK ? mrow : 0) * lda;
  const bool aVec = ((lda & 3) == 0);

  // TRANSB per-lane row pointers (clamped)
  const int  nT0 = tn + ll, nT1 = tn + 16 + ll;
  const float* Brow0 = B + (long long)((nT0 < N) ? nT0 : 0) * ldb;
  const float* Brow1 = B + (long long)((nT1 < N) ? nT1 : 0) * ldb;
  const bool bVec = ((ldb & 3) == 0);

  for (int k0 = 0; k0 < K; k0 += KCHUNK) {
    const int rows = (K - k0 < KCHUNK) ? (K - k0) : KCHUNK;

    if (!(flags & GF_TRANSB)) {
#if __has_builtin(__builtin_amdgcn_tensor_load_to_lds)
      // --- Tensor Data Mover: stage B[k0 .. k0+rows, tn .. tn+32) into LDS ---
      if (wv == 0) {
        unsigned long long gaddr =
            (unsigned long long)(const void*)(B + (long long)k0 * ldb + tn);
        unsigned int ldsoff = (unsigned int)(unsigned long long)(void*)&smemB[0];
        unsigned int td0 = (unsigned int)(N - tn);        // cols remaining (OOB->0)
        unsigned int td1 = (unsigned int)rows;
        unsigned int tile0 = 32u;
        unsigned int tile1 = (unsigned int)rows;
        unsigned long long st0 = (unsigned long long)ldb; // row stride (elements)
        // D# group 0: count=1 | lds_addr | global_addr(57b) | type=2
        v4u g0;
        g0[0] = 1u;
        g0[1] = ldsoff;
        g0[2] = (unsigned int)(gaddr & 0xFFFFFFFFull);
        g0[3] = (unsigned int)((gaddr >> 32) & 0x01FFFFFFull) | 0x80000000u;
        // D# group 1: data_size=4B (code 2), tensor dims, tile dims, stride0
        v8i g1;
        g1[0] = (int)(2u << 16);
        g1[1] = (int)((td0 & 0xFFFFu) << 16);
        g1[2] = (int)((td0 >> 16) | ((td1 & 0xFFFFu) << 16));
        g1[3] = (int)((td1 >> 16) | (tile0 << 16));
        g1[4] = (int)(tile1 & 0xFFFFu);
        g1[5] = (int)(st0 & 0xFFFFFFFFull);
        g1[6] = (int)((st0 >> 32) & 0xFFFFull);
        g1[7] = 0;
        v4i gz = {0, 0, 0, 0};
#if __clang_major__ >= 23
        v8i gz8 = {0, 0, 0, 0, 0, 0, 0, 0};
        __builtin_amdgcn_tensor_load_to_lds(g0, g1, gz, gz, gz8, 0);
#else
        __builtin_amdgcn_tensor_load_to_lds(g0, g1, gz, gz, 0);
#endif
        __builtin_amdgcn_s_wait_tensorcnt(0);
      }
      __syncthreads();
#else
      // Fallback: cooperative staged load
      for (int i = threadIdx.x; i < rows * 32; i += 128) {
        int y = i >> 5, x = i & 31;
        float v = 0.f;
        int n = tn + x;
        if (n < N) v = B[(long long)(k0 + y) * ldb + n];
        smemB[i] = v;
      }
      __syncthreads();
#endif
    }

    // prefetch next A panel
    if (k0 + KCHUNK < K && rowOK)
      __builtin_prefetch(&Arow[k0 + KCHUNK], 0, 1);

    for (int kk = 0; kk < rows; kk += 32) {
      const int kb = k0 + kk;
      const bool kfull = (kb + 31 < K);

      // ---- A fragment (16x32 f16); lane row = ll, K split contiguous (k = 16*lh + e) ----
      v16h af;
      if (kfull && aVec) {
        const float4* p0 = (const float4*)(Arow + kb + lh * 16);
        float4 x0 = p0[0], x1 = p0[1], x2 = p0[2], x3 = p0[3]; // +0,+4,+8,+12
        if (!rowOK) { x0 = make_float4(0,0,0,0); x1 = x0; x2 = x0; x3 = x0; }
        af[0]=(_Float16)x0.x; af[1]=(_Float16)x0.y; af[2]=(_Float16)x0.z; af[3]=(_Float16)x0.w;
        af[4]=(_Float16)x1.x; af[5]=(_Float16)x1.y; af[6]=(_Float16)x1.z; af[7]=(_Float16)x1.w;
        af[8]=(_Float16)x2.x; af[9]=(_Float16)x2.y; af[10]=(_Float16)x2.z; af[11]=(_Float16)x2.w;
        af[12]=(_Float16)x3.x; af[13]=(_Float16)x3.y; af[14]=(_Float16)x3.z; af[15]=(_Float16)x3.w;
      } else {
#pragma unroll
        for (int e = 0; e < 16; ++e) {
          int k  = kb + lh * 16 + e;
          int kc = (k < K) ? k : 0;
          float v = Arow[kc];                      // unconditional (clamped) load
          v = (k < K && rowOK) ? v : 0.f;          // select, no branch
          af[e] = (_Float16)v;
        }
      }

      // ---- B fragments (32x16 f16 each); lane col = ll / ll+16 --------------
      v16h bf0, bf1;
      if (flags & GF_TRANSB) {
        if (kfull && bVec) {
          const float4* q0 = (const float4*)(Brow0 + kb + lh * 16);
          const float4* q1 = (const float4*)(Brow1 + kb + lh * 16);
          float4 y0 = q0[0], y1 = q0[1], y2 = q0[2], y3 = q0[3];
          float4 z0 = q1[0], z1 = q1[1], z2 = q1[2], z3 = q1[3];
          if (nT0 >= N) { y0 = make_float4(0,0,0,0); y1 = y0; y2 = y0; y3 = y0; }
          if (nT1 >= N) { z0 = make_float4(0,0,0,0); z1 = z0; z2 = z0; z3 = z0; }
          bf0[0]=(_Float16)y0.x; bf0[1]=(_Float16)y0.y; bf0[2]=(_Float16)y0.z; bf0[3]=(_Float16)y0.w;
          bf0[4]=(_Float16)y1.x; bf0[5]=(_Float16)y1.y; bf0[6]=(_Float16)y1.z; bf0[7]=(_Float16)y1.w;
          bf0[8]=(_Float16)y2.x; bf0[9]=(_Float16)y2.y; bf0[10]=(_Float16)y2.z; bf0[11]=(_Float16)y2.w;
          bf0[12]=(_Float16)y3.x; bf0[13]=(_Float16)y3.y; bf0[14]=(_Float16)y3.z; bf0[15]=(_Float16)y3.w;
          bf1[0]=(_Float16)z0.x; bf1[1]=(_Float16)z0.y; bf1[2]=(_Float16)z0.z; bf1[3]=(_Float16)z0.w;
          bf1[4]=(_Float16)z1.x; bf1[5]=(_Float16)z1.y; bf1[6]=(_Float16)z1.z; bf1[7]=(_Float16)z1.w;
          bf1[8]=(_Float16)z2.x; bf1[9]=(_Float16)z2.y; bf1[10]=(_Float16)z2.z; bf1[11]=(_Float16)z2.w;
          bf1[12]=(_Float16)z3.x; bf1[13]=(_Float16)z3.y; bf1[14]=(_Float16)z3.z; bf1[15]=(_Float16)z3.w;
        } else {
#pragma unroll
          for (int e = 0; e < 16; ++e) {
            int k  = kb + lh * 16 + e;
            int kc = (k < K) ? k : 0;
            float v0 = Brow0[kc];
            float v1 = Brow1[kc];
            v0 = (k < K && nT0 < N) ? v0 : 0.f;
            v1 = (k < K && nT1 < N) ? v1 : 0.f;
            bf0[e] = (_Float16)v0;
            bf1[e] = (_Float16)v1;
          }
        }
      } else {
#pragma unroll
        for (int e = 0; e < 16; ++e) {
          int kL = kk + lh * 16 + e;
          int kc = (kL < rows) ? kL : 0;
          float v0 = smemB[kc * 32 + ll];
          float v1 = smemB[kc * 32 + 16 + ll];
          v0 = (kL < rows) ? v0 : 0.f;
          v1 = (kL < rows) ? v1 : 0.f;
          bf0[e] = (_Float16)v0;
          bf1[e] = (_Float16)v1;
        }
      }

      acc0 = __builtin_amdgcn_wmma_f32_16x16x32_f16(
          false, af, false, bf0, (short)0, acc0, false, false);
      acc1 = __builtin_amdgcn_wmma_f32_16x16x32_f16(
          false, af, false, bf1, (short)0, acc1, false, false);
    }

    if (!(flags & GF_TRANSB)) __syncthreads();
  }

  // ---- fused epilogue + store (C/D layout: n = ll, m = tm + r + 8*lh) ------
  auto epi = [&](float v, int m, int n) -> float {
    if (flags & GF_COLBIAS) v += cb[n];
    if (flags & GF_RELU)    v = fmaxf(v, 0.f);
    if (flags & GF_GELU)    v = 0.5f * v * (1.f + erff(v * 0.70710678f));
    if (flags & GF_ADD)     v += addend[(long long)m * ldadd + n];
    return v;
  };
  const int n0 = tn + ll, n1 = tn + 16 + ll;
#pragma unroll
  for (int r = 0; r < 8; ++r) {
    int m = tm + r + 8 * lh;
    if (m < M) {
      float gsc = (flags & GF_ROWSCALE) ? rs[m] : 1.f;
      float rbi = (flags & GF_ROWBIAS)  ? rb[m] : 0.f;
      if (n0 < N) C[(long long)m * ldc + n0] = epi(acc0[r] * alpha * gsc + rbi, m, n0);
      if (n1 < N) C[(long long)m * ldc + n1] = epi(acc1[r] * alpha * gsc + rbi, m, n1);
    }
  }
}

// ---------------------------------------------------------------------------
// Assemble xyzs (B,3,N,3) + feats (B,3,3,N) from inp (B,N,6) and vid (B,2,N,6)
// ---------------------------------------------------------------------------
__global__ void assemble_kernel(const float* __restrict__ inp, const float* __restrict__ vid,
                                float* __restrict__ xyzs, float* __restrict__ feats,
                                int N, int total)
{
  int i = blockIdx.x * 256 + threadIdx.x;
  if (i >= total) return;
  int n = i % N; int r = i / N; int l = r % 3; int b = r / 3;
  const float* src = (l == 0) ? (inp + ((long long)b * N + n) * 6)
                              : (vid + ((((long long)b * 2 + (l - 1)) * N) + n) * 6);
  float* xo = xyzs + (((long long)(b * 3 + l) * N) + n) * 3;
  xo[0] = src[0]; xo[1] = src[1]; xo[2] = src[2];
  float* fo = feats + ((long long)(b * 3 + l) * 3) * (long long)N;
  fo[0 * (long long)N + n] = src[3];
  fo[1 * (long long)N + n] = src[4];
  fo[2 * (long long)N + n] = src[5];
}

// ---------------------------------------------------------------------------
// Farthest point sampling: one block per (b, frame); mind kept in workspace.
// ---------------------------------------------------------------------------
__global__ void fps_kernel(const float* __restrict__ xyz, int* __restrict__ idx_out,
                           float* __restrict__ mind_ws, int N, int M)
{
  int blk = blockIdx.x;
  const float* p = xyz + (long long)blk * N * 3;
  int* out = idx_out + (long long)blk * M;
  float* mind = mind_ws + (long long)blk * N;
  __shared__ float s_val[256];
  __shared__ int   s_idx[256];
  int tid = threadIdx.x;
  for (int i = tid; i < N; i += 256) mind[i] = 1e10f;
  if (tid == 0) out[0] = 0;
  __syncthreads();
  int last = 0;
  for (int it = 1; it < M; ++it) {
    float lx = p[last * 3], ly = p[last * 3 + 1], lz = p[last * 3 + 2];
    float best = -1.f; int bi = 0;
    for (int i = tid; i < N; i += 256) {
      float dx = p[i * 3] - lx, dy = p[i * 3 + 1] - ly, dz = p[i * 3 + 2] - lz;
      float d = dx * dx + dy * dy + dz * dz;
      float m = fminf(mind[i], d);
      mind[i] = m;
      if (m > best) { best = m; bi = i; }
    }
    s_val[tid] = best; s_idx[tid] = bi;
    __syncthreads();
    for (int s = 128; s > 0; s >>= 1) {
      if (tid < s && s_val[tid + s] > s_val[tid]) { s_val[tid] = s_val[tid + s]; s_idx[tid] = s_idx[tid + s]; }
      __syncthreads();
    }
    last = s_idx[0];
    if (tid == 0) out[it] = last;
    __syncthreads();
  }
}

__global__ void gather_xyz_kernel(const float* __restrict__ xyz, const int* __restrict__ idx,
                                  float* __restrict__ out, int N, int M, int total)
{
  int i = blockIdx.x * 256 + threadIdx.x;
  if (i >= total) return;
  int m = i % M; int bl = i / M;
  int id = idx[(long long)bl * M + m];
  const float* s = xyz + ((long long)bl * N + id) * 3;
  float* d = out + ((long long)bl * M + m) * 3;
  d[0] = s[0]; d[1] = s[1]; d[2] = s[2];
}

// ---------------------------------------------------------------------------
// Ball query: one wave per anchor. Selects the first 32 indices (ascending)
// within radius via wave32 ballot + prefix popcount (== reference top_k trick).
// ---------------------------------------------------------------------------
__global__ void ball_query_kernel(const float* __restrict__ pts, long long ptsBS,
                                  const float* __restrict__ anc, long long ancBS,
                                  int* __restrict__ idx, int N, int M, float r2, int Bn)
{
  int w = blockIdx.x * (blockDim.x >> 5) + (threadIdx.x >> 5);
  int lane = threadIdx.x & 31;
  if (w >= Bn * M) return;
  int m = w % M, b = w / M;
  const float* p = pts + (long long)b * ptsBS;
  const float* a = anc + (long long)b * ancBS + (long long)m * 3;
  float ax = a[0], ay = a[1], az = a[2];
  int* out = idx + ((long long)b * M + m) * 32;
  int cnt = 0, first = 0, haveFirst = 0;
  for (int base = 0; base < N && cnt < 32; base += 32) {
    int i0 = base + lane;
    bool ok = false;
    if (i0 < N) {
      float dx = p[i0 * 3] - ax, dy = p[i0 * 3 + 1] - ay, dz = p[i0 * 3 + 2] - az;
      ok = (dx * dx + dy * dy + dz * dz) < r2;
    }
    unsigned msk = (unsigned)__ballot(ok);
    if (!haveFirst && msk) { first = base + (__ffs(msk) - 1); haveFirst = 1; }
    int pre = __popc(msk & ((1u << lane) - 1u));
    if (ok && cnt + pre < 32) out[cnt + pre] = i0;
    cnt += __popc(msk);
  }
  if (cnt > 32) cnt = 32;
  for (int j = cnt + lane; j < 32; j += 32) out[j] = first;
}

// Displacement (+temporal channel) grouping: D (B, 4, M*32)
__global__ void group_disp_kernel(const float* __restrict__ pts, long long ptsBS,
                                  const float* __restrict__ anc, long long ancBS,
                                  const int* __restrict__ idx, float* __restrict__ D,
                                  int M, float tval, int total)
{
  int i = blockIdx.x * 256 + threadIdx.x;
  if (i >= total) return;
  int j = i & 31; int m = (i >> 5) % M; int b = i / (32 * M);
  int id = idx[((long long)b * M + m) * 32 + j];
  const float* p = pts + (long long)b * ptsBS + (long long)id * 3;
  const float* a = anc + (long long)b * ancBS + (long long)m * 3;
  long long s = (long long)M * 32;
  float* d = D + (long long)b * 4 * s + ((long long)m * 32 + j);
  d[0]     = p[0] - a[0];
  d[s]     = p[1] - a[1];
  d[2 * s] = p[2] - a[2];
  d[3 * s] = tval;
}

// Feature grouping: G (B, Cin, M*32)
__global__ void group_feats_kernel(const float* __restrict__ f, long long fBS,
                                   const int* __restrict__ idx, float* __restrict__ G,
                                   int N, int M, int Cin, int total)
{
  int i = blockIdx.x * 256 + threadIdx.x;
  if (i >= total) return;
  int j = i & 31; int r = i >> 5; int m = r % M; r /= M; int c = r % Cin; int b = r / Cin;
  int id = idx[((long long)b * M + m) * 32 + j];
  G[(((long long)b * Cin + c) * M * 32) + (long long)m * 32 + j] =
      f[(long long)b * fBS + (long long)c * N + id];
}

// Max over the 32 samples, with optional accumulation across temporal window.
__global__ void maxpool_kernel(const float* __restrict__ X, float* __restrict__ out,
                               long long outBS, int C, int M, int accum, int total)
{
  int i = blockIdx.x * 256 + threadIdx.x;
  if (i >= total) return;
  int m = i % M; int r = i / M; int c = r % C; int b = r / C;
  const float* xp = X + (((long long)b * C + c) * M + m) * 32;
  float mx = xp[0];
#pragma unroll 4
  for (int j = 1; j < 32; ++j) mx = fmaxf(mx, xp[j]);
  float* o = out + (long long)b * outBS + (long long)c * M + m;
  if (accum) *o += mx; else *o = mx;
}

__global__ void layernorm_kernel(const float* __restrict__ x, float* __restrict__ y,
                                 const float* __restrict__ g, const float* __restrict__ b,
                                 int rows, int dim)
{
  int row = blockIdx.x * (blockDim.x >> 5) + (threadIdx.x >> 5);
  if (row >= rows) return;
  int lane = threadIdx.x & 31;
  const float* xr = x + (long long)row * dim;
  float* yr = y + (long long)row * dim;
  float s = 0.f, s2 = 0.f;
  for (int c = lane; c < dim; c += 32) { float v = xr[c]; s += v; s2 += v * v; }
  for (int o = 16; o > 0; o >>= 1) { s += __shfl_down(s, o, 32); s2 += __shfl_down(s2, o, 32); }
  s = __shfl(s, 0, 32); s2 = __shfl(s2, 0, 32);
  float mean = s / dim;
  float var = s2 / dim - mean * mean;
  float inv = rsqrtf(var + 1e-5f);
  for (int c = lane; c < dim; c += 32) yr[c] = (xr[c] - mean) * inv * g[c] + b[c];
}

__global__ void softmax_kernel(float* __restrict__ x, int rows, int dim)
{
  int row = blockIdx.x * (blockDim.x >> 5) + (threadIdx.x >> 5);
  if (row >= rows) return;
  int lane = threadIdx.x & 31;
  float* xr = x + (long long)row * dim;
  float mx = -1e30f;
  for (int c = lane; c < dim; c += 32) mx = fmaxf(mx, xr[c]);
  for (int o = 16; o > 0; o >>= 1) mx = fmaxf(mx, __shfl_down(mx, o, 32));
  mx = __shfl(mx, 0, 32);
  float s = 0.f;
  for (int c = lane; c < dim; c += 32) { float e = __expf(xr[c] - mx); xr[c] = e; s += e; }
  for (int o = 16; o > 0; o >>= 1) s += __shfl_down(s, o, 32);
  s = __shfl(s, 0, 32);
  float inv = 1.f / s;
  for (int c = lane; c < dim; c += 32) xr[c] *= inv;
}

// f4 (B,3,256,32) -> emb tokens (B,96,256) with ReLU
__global__ void emb_kernel(const float* __restrict__ f4, float* __restrict__ x, int total)
{
  int i = blockIdx.x * 256 + threadIdx.x;
  if (i >= total) return;
  int n = i & 31; int r = i >> 5; int c = r & 255; r >>= 8; int l = r % 3; int b = r / 3;
  x[(((long long)b * 96 + l * 32 + n) * 256) + c] = fmaxf(f4[i], 0.f);
}

// tokens (B,96,256) -> f4t (B,3,256,32)
__global__ void f4t_kernel(const float* __restrict__ x, float* __restrict__ f4t, int total)
{
  int i = blockIdx.x * 256 + threadIdx.x;
  if (i >= total) return;
  int n = i & 31; int r = i >> 5; int c = r & 255; r >>= 8; int l = r % 3; int b = r / 3;
  f4t[i] = x[(((long long)b * 96 + l * 32 + n) * 256) + c];
}

// Copy C rows into a concat buffer at row offset rowOff.
__global__ void copy_rows_kernel(const float* __restrict__ src, long long sStride,
                                 float* __restrict__ dst, long long dStride,
                                 int rowOff, int C, int Np, int total)
{
  int i = blockIdx.x * 256 + threadIdx.x;
  if (i >= total) return;
  int n = i % Np; int r = i / Np; int c = r % C; int bt = r / C;
  dst[(long long)bt * dStride + (long long)(rowOff + c) * Np + n] =
      src[(long long)bt * sStride + (long long)c * Np + n];
}

// 3-NN inverse-distance interpolation; one wave per output point.
__global__ void three_interp_kernel(const float* __restrict__ feats, long long fStride,
                                    const float* __restrict__ kxyz, long long kStride,
                                    const float* __restrict__ uxyz, long long uStride,
                                    float* __restrict__ out, long long oStride,
                                    int C, int Nk, int Nu, int ldOut, int totalW)
{
  int w = blockIdx.x * (blockDim.x >> 5) + (threadIdx.x >> 5);
  int lane = threadIdx.x & 31;
  if (w >= totalW) return;
  int u = w % Nu; int bt = w / Nu;
  const float* up = uxyz + (long long)bt * uStride + (long long)u * 3;
  float ux = up[0], uy = up[1], uz = up[2];
  float d0 = 1e30f, d1 = 1e30f, d2 = 1e30f;
  int i0 = 0, i1 = 0, i2 = 0;
  if (lane == 0) {
    const float* kb = kxyz + (long long)bt * kStride;
    for (int j = 0; j < Nk; ++j) {
      float dx = ux - kb[j * 3], dy = uy - kb[j * 3 + 1], dz = uz - kb[j * 3 + 2];
      float d = dx * dx + dy * dy + dz * dz;
      if (d < d0)      { d2 = d1; i2 = i1; d1 = d0; i1 = i0; d0 = d; i0 = j; }
      else if (d < d1) { d2 = d1; i2 = i1; d1 = d;  i1 = j; }
      else if (d < d2) { d2 = d;  i2 = j; }
    }
  }
  i0 = __shfl(i0, 0, 32); i1 = __shfl(i1, 0, 32); i2 = __shfl(i2, 0, 32);
  d0 = __shfl(d0, 0, 32); d1 = __shfl(d1, 0, 32); d2 = __shfl(d2, 0, 32);
  float w0 = 1.f / (sqrtf(fmaxf(d0, 1e-12f)) + 1e-8f);
  float w1 = 1.f / (sqrtf(fmaxf(d1, 1e-12f)) + 1e-8f);
  float w2 = 1.f / (sqrtf(fmaxf(d2, 1e-12f)) + 1e-8f);
  float ws = w0 + w1 + w2;
  w0 /= ws; w1 /= ws; w2 /= ws;
  const float* f = feats + (long long)bt * fStride;
  float* o = out + (long long)bt * oStride;
  for (int c = lane; c < C; c += 32)
    o[(long long)c * ldOut + u] =
        w0 * f[(long long)c * Nk + i0] + w1 * f[(long long)c * Nk + i1] + w2 * f[(long long)c * Nk + i2];
}

// out[b,n,o]: o<288 -> sum_l w[l]*fd3[b,l,o,n] + bias ; o>=288 -> x2 frame0 coords
__global__ void final_kernel(const float* __restrict__ fd3, const float* __restrict__ x2xyz,
                             const float* __restrict__ wout, const float* __restrict__ bout,
                             float* __restrict__ out, int total)
{
  int i = blockIdx.x * 256 + threadIdx.x;
  if (i >= total) return;
  int o = i % 291; int r = i / 291; int n = r % 128; int b = r / 128;
  float v;
  if (o < 288) {
    v = bout[0];
#pragma unroll
    for (int l = 0; l < 3; ++l)
      v += wout[l] * fd3[((((long long)b * 3 + l) * 288) + o) * 128 + n];
  } else {
    v = x2xyz[(((long long)b * 3 + 0) * 128 + n) * 3 + (o - 288)];
  }
  out[i] = v;
}

// ---------------------------------------------------------------------------
extern "C" void kernel_launch(void* const* d_in, const int* in_sizes, int n_in,
                              void* d_out, int out_size, void* d_ws, size_t ws_size,
                              hipStream_t stream)
{
  (void)n_in; (void)out_size; (void)ws_size;
  auto F = [&](int i) { return (const float*)d_in[i]; };
  const int B  = in_sizes[0] / (25600 * 6);
  const int Bt = B * 3;

  // ---- workspace bump allocator -------------------------------------------
  size_t off = 0;
  auto alloc  = [&](long long n) { float* p = (float*)((char*)d_ws + off); off += (size_t)((n * 4 + 255) / 256 * 256); return p; };
  auto alloci = [&](long long n) { int*   p = (int*)  ((char*)d_ws + off); off += (size_t)((n * 4 + 255) / 256 * 256); return p; };

  float* xyzs   = alloc((long long)Bt * 25600 * 3);
  float* feats0 = alloc((long long)Bt * 3 * 25600);
  float* mindws = alloc((long long)Bt * 25600);
  int*   fpsidx = alloci((long long)Bt * 256);
  int*   ballidx= alloci((long long)B * 256 * 32);
  float* bufD   = alloc((long long)B * 4 * 8192);
  float* bufG   = alloc((long long)B * 524288);
  float* dbuf   = alloc((long long)B * 524288);
  float* fbA    = alloc((long long)B * 1048576);
  float* fbB    = alloc((long long)B * 1048576);
  float* sx[5]; float* sf[5];
  sx[0] = xyzs; sf[0] = feats0;
  sx[1] = alloc((long long)Bt * 256 * 3); sf[1] = alloc((long long)Bt * 128 * 256);
  sx[2] = alloc((long long)Bt * 128 * 3); sf[2] = alloc((long long)Bt * 256 * 128);
  sx[3] = alloc((long long)Bt * 64 * 3);  sf[3] = alloc((long long)Bt * 256 * 64);
  sx[4] = alloc((long long)Bt * 32 * 3);  sf[4] = alloc((long long)Bt * 256 * 32);
  float* xbA  = alloc((long long)B * 96 * 256);
  float* xbB  = alloc((long long)B * 96 * 256);
  float* ybuf = alloc((long long)B * 96 * 256);
  float* qkv  = alloc((long long)B * 96 * 3072);
  float* scor = alloc((long long)B * 4 * 96 * 96);
  float* obuf = alloc((long long)B * 96 * 1024);
  float* ffb  = alloc((long long)B * 96 * 256);
  float* f4t  = alloc((long long)Bt * 256 * 32);
  float* cat  = alloc((long long)Bt * 512 * 128);
  float* tb1  = alloc((long long)Bt * 256 * 128);
  float* fd4  = alloc((long long)Bt * 256 * 64);
  float* fd3  = alloc((long long)Bt * 288 * 128);

  auto gemm = [&](const float* A, const float* Bm, float* C,
                  const float* rs, const float* rb, const float* cb, const float* add,
                  int M, int N, int K, int lda, int ldb, int ldc, int ldadd,
                  long long sA, long long sB, long long sC, long long sAdd,
                  float alpha, int flags, int batch) {
    dim3 g((N + 31) / 32, (M + 63) / 64, batch);
    gemm_wmma_kernel<<<g, dim3(128), 0, stream>>>(A, Bm, C, rs, rb, cb, add,
        M, N, K, lda, ldb, ldc, ldadd, sA, sB, sC, sAdd, alpha, flags);
  };

  // ---- assemble point clouds ----------------------------------------------
  {
    int tot = Bt * 25600;
    assemble_kernel<<<(tot + 255) / 256, 256, 0, stream>>>(F(0), F(1), xyzs, feats0, 25600, tot);
  }

  // ---- p4dconv stages ------------------------------------------------------
  struct StageCfg { int Nin, M, cin, c0, c1, c2, tk; float r; int pb; };
  const StageCfg st[4] = {
    {25600, 256,   3,  32,  64, 128, 1, 0.9f,  2},
    {  256, 128, 128, 128, 128, 256, 1, 1.8f, 14},
    {  128,  64, 256, 256, 256, 256, 3, 3.6f, 26},
    {   64,  32, 256, 256, 256, 256, 1, 7.2f, 38},
  };

  for (int s = 0; s < 4; ++s) {
    const StageCfg c = st[s];
    const float* Wd = F(c.pb + 0); const float* gd = F(c.pb + 1); const float* bd = F(c.pb + 2);
    const float* Wf = F(c.pb + 3); const float* gf = F(c.pb + 4); const float* bf = F(c.pb + 5);
    const float* W1 = F(c.pb + 6); const float* g1 = F(c.pb + 7); const float* b1 = F(c.pb + 8);
    const float* W2 = F(c.pb + 9); const float* g2 = F(c.pb +10); const float* b2 = F(c.pb +11);
    const float* xin = sx[s]; const float* fin = sf[s];
    float* xo = sx[s + 1]; float* fo = sf[s + 1];
    const int Nin = c.Nin, M = c.M, cin = c.cin, pos = M * 32;
    const long long xinBS = 3LL * Nin * 3, xoBS = 3LL * M * 3, finBS = 3LL * cin * Nin;

    fps_kernel<<<Bt, 256, 0, stream>>>(xin, fpsidx, mindws, Nin, M);
    { int tot = Bt * M;
      gather_xyz_kernel<<<(tot + 255) / 256, 256, 0, stream>>>(xin, fpsidx, xo, Nin, M, tot); }

    for (int t = 0; t < 3; ++t) {
      const int nwin = (c.tk == 3) ? 3 : 1;
      for (int wi = 0; wi < nwin; ++wi) {
        int dt = (c.tk == 3) ? (wi - 1) : 0;
        int src = t + dt; src = src < 0 ? 0 : (src > 2 ? 2 : src);
        { int tot = B * M;
          ball_query_kernel<<<(tot + 3) / 4, 128, 0, stream>>>(
              xin + (long long)src * Nin * 3, xinBS, xo + (long long)t * M * 3, xoBS,
              ballidx, Nin, M, c.r * c.r, B); }
        { int tot = B * M * 32;
          group_disp_kernel<<<(tot + 255) / 256, 256, 0, stream>>>(
              xin + (long long)src * Nin * 3, xinBS, xo + (long long)t * M * 3, xoBS,
              ballidx, bufD, M, (float)dt, tot); }
        { int tot = B * cin * M * 32;
          group_feats_kernel<<<(tot + 255) / 256, 256, 0, stream>>>(
              fin + (long long)src * cin * Nin, finBS, ballidx, bufG, Nin, M, cin, tot); }
        gemm(Wd, bufD, dbuf, gd, bd, nullptr, nullptr, c.c0, pos, 4,
             4, pos, pos, 0, 0, 4LL * pos, (long long)c.c0 * pos, 0,
             1.f, GF_RELU | GF_ROWSCALE | GF_ROWBIAS, B);
        gemm(Wf, bufG, fbA, gf, bf, nullptr, dbuf, c.c0, pos, cin,
             cin, pos, pos, pos, 0, (long long)cin * pos, (long long)c.c0 * pos, (long long)c.c0 * pos,
             1.f, GF_RELU | GF_ROWSCALE | GF_ROWBIAS | GF_ADD, B);
        gemm(W1, fbA, fbB, g1, b1, nullptr, nullptr, c.c1, pos, c.c0,
             c.c0, pos, pos, 0, 0, (long long)c.c0 * pos, (long long)c.c1 * pos, 0,
             1.f, GF_RELU | GF_ROWSCALE | GF_ROWBIAS, B);
        gemm(W2, fbB, fbA, g2, b2, nullptr, nullptr, c.c2, pos, c.c1,
             c.c1, pos, pos, 0, 0, (long long)c.c1 * pos, (long long)c.c2 * pos, 0,
             1.f, GF_RELU | GF_ROWSCALE | GF_ROWBIAS, B);
        { int tot = B * c.c2 * M;
          maxpool_kernel<<<(tot + 255) / 256, 256, 0, stream>>>(
              fbA, fo + (long long)t * c.c2 * M, 3LL * c.c2 * M, c.c2, M, (wi > 0) ? 1 : 0, tot); }
      }
    }
  }

  // ---- transformer (depth 1, h=4, dh=256) ---------------------------------
  { int tot = Bt * 256 * 32;
    emb_kernel<<<(tot + 255) / 256, 256, 0, stream>>>(sf[4], xbA, tot); }
  layernorm_kernel<<<(B * 96 + 7) / 8, 256, 0, stream>>>(xbA, ybuf, F(50), F(51), B * 96, 256);
  gemm(ybuf, F(52), qkv, nullptr, nullptr, nullptr, nullptr, 96, 3072, 256,
       256, 3072, 3072, 0, 96LL * 256, 0, 96LL * 3072, 0, 1.f, 0, B);
  for (int h = 0; h < 4; ++h)
    gemm(qkv + h * 256, qkv + 1024 + h * 256, scor + h * 96 * 96, nullptr, nullptr, nullptr, nullptr,
         96, 96, 256, 3072, 3072, 96, 0, 96LL * 3072, 96LL * 3072, 4LL * 96 * 96, 0,
         0.0625f, GF_TRANSB, B);
  softmax_kernel<<<(B * 4 * 96 + 7) / 8, 256, 0, stream>>>(scor, B * 4 * 96, 96);
  for (int h = 0; h < 4; ++h)
    gemm(scor + h * 96 * 96, qkv + 2048 + h * 256, obuf + h * 256, nullptr, nullptr, nullptr, nullptr,
         96, 256, 96, 96, 3072, 1024, 0, 4LL * 96 * 96, 96LL * 3072, 96LL * 1024, 0,
         1.f, 0, B);
  gemm(obuf, F(53), xbB, nullptr, nullptr, F(54), xbA, 96, 256, 1024,
       1024, 256, 256, 256, 96LL * 1024, 0, 96LL * 256, 96LL * 256,
       1.f, GF_COLBIAS | GF_ADD, B);
  layernorm_kernel<<<(B * 96 + 7) / 8, 256, 0, stream>>>(xbB, ybuf, F(55), F(56), B * 96, 256);
  gemm(ybuf, F(57), ffb, nullptr, nullptr, F(58), nullptr, 96, 256, 256,
       256, 256, 256, 0, 96LL * 256, 0, 96LL * 256, 0, 1.f, GF_COLBIAS | GF_GELU, B);
  gemm(ffb, F(59), xbA, nullptr, nullptr, F(60), xbB, 96, 256, 256,
       256, 256, 256, 256, 96LL * 256, 0, 96LL * 256, 96LL * 256, 1.f, GF_COLBIAS | GF_ADD, B);
  { int tot = Bt * 256 * 32;
    f4t_kernel<<<(tot + 255) / 256, 256, 0, stream>>>(xbA, f4t, tot); }

  // ---- deconv4: x4(32) -> x3(64) ------------------------------------------
  { int tot = Bt * 64;
    three_interp_kernel<<<(tot + 7) / 8, 256, 0, stream>>>(
        f4t, 256LL * 32, sx[4], 32LL * 3, sx[3], 64LL * 3, cat, 512LL * 64,
        256, 32, 64, 64, tot); }
  { int tot = Bt * 256 * 64;
    copy_rows_kernel<<<(tot + 255) / 256, 256, 0, stream>>>(sf[3], 256LL * 64, cat, 512LL * 64,
                                                            256, 256, 64, tot); }
  gemm(F(61), cat, tb1, F(62), F(63), nullptr, nullptr, 256, 64, 512,
       512, 64, 64, 0, 0, 512LL * 64, 256LL * 64, 0, 1.f, GF_RELU | GF_ROWSCALE | GF_ROWBIAS, Bt);
  gemm(F(64), tb1, fd4, F(65), F(66), nullptr, nullptr, 256, 64, 256,
       256, 64, 64, 0, 0, 256LL * 64, 256LL * 64, 0, 1.f, GF_RELU | GF_ROWSCALE | GF_ROWBIAS, Bt);

  // ---- deconv3: x3(64) -> x2(128) -----------------------------------------
  { int tot = Bt * 128;
    three_interp_kernel<<<(tot + 7) / 8, 256, 0, stream>>>(
        fd4, 256LL * 64, sx[3], 64LL * 3, sx[2], 128LL * 3, cat, 512LL * 128,
        256, 64, 128, 128, tot); }
  { int tot = Bt * 256 * 128;
    copy_rows_kernel<<<(tot + 255) / 256, 256, 0, stream>>>(sf[2], 256LL * 128, cat, 512LL * 128,
                                                            256, 256, 128, tot); }
  gemm(F(67), cat, tb1, F(68), F(69), nullptr, nullptr, 256, 128, 512,
       512, 128, 128, 0, 0, 512LL * 128, 256LL * 128, 0, 1.f, GF_RELU | GF_ROWSCALE | GF_ROWBIAS, Bt);
  gemm(F(70), tb1, fd3, F(71), F(72), nullptr, nullptr, 288, 128, 256,
       256, 128, 128, 0, 0, 256LL * 128, 288LL * 128, 0, 1.f, GF_RELU | GF_ROWSCALE | GF_ROWBIAS, Bt);

  // ---- output head ---------------------------------------------------------
  { int tot = B * 128 * 291;
    final_kernel<<<(tot + 255) / 256, 256, 0, stream>>>(fd3, sx[2], F(73), F(74),
                                                        (float*)d_out, tot); }
}
